// GroupedQueryAttention_1322849927596
// MI455X (gfx1250) — compile-verified, matrix operand order checked by reading, not run on hardware
//
#include <hip/hip_runtime.h>

// ---------------- problem constants ----------------
constexpr int kB   = 2;
constexpr int kS   = 2048;
constexpr int kH   = 2304;   // NH*HD
constexpr int kNH  = 9;
constexpr int kNKV = 3;
constexpr int kHD  = 256;
constexpr int kGR  = kNH / kNKV;      // 3
constexpr int kM   = kB * kS;         // 4096 rows
constexpr float kScale = 0.0625f;     // 1/sqrt(256)
constexpr float kLnBase = 11.512925464970229f; // ln(1e5)

// ---------------- CDNA5 async / TDM availability ----------------
#define HAS_ASYNC (__has_builtin(__builtin_amdgcn_global_load_async_to_lds_b128) && \
                   __has_builtin(__builtin_amdgcn_s_wait_asynccnt))
#define HAS_TDM   (__has_builtin(__builtin_amdgcn_tensor_load_to_lds) && \
                   __has_builtin(__builtin_amdgcn_s_wait_tensorcnt))

// async builtin signature (from hipcc diagnostic): (int4 AS1*, int4 AS3*, imm, imm)
typedef int v4i __attribute__((vector_size(16)));
typedef __attribute__((address_space(1))) v4i as1_v4i;
typedef __attribute__((address_space(3))) v4i as3_v4i;
typedef __attribute__((address_space(3))) unsigned short as3_u16;

// TDM descriptor groups (this toolchain: 6-arg form u32x4, i32x8, i32x4, i32x4, i32x8, i32)
typedef unsigned tdm_u32x4 __attribute__((ext_vector_type(4)));
typedef int      tdm_i32x8 __attribute__((ext_vector_type(8)));
typedef int      tdm_i32x4 __attribute__((ext_vector_type(4)));

// ---------------- WMMA types ----------------
typedef __attribute__((ext_vector_type(16))) __bf16 v16bf;
typedef __attribute__((ext_vector_type(8)))  float  v8f;

union FragAB {
    v16bf v;
    uint4 q[2];
    unsigned u[8];
    __bf16 b[16];
};
union FragC {
    v8f v;
    float f[8];
};

__device__ __forceinline__ unsigned short f2bf_u16(float f) {
    union { float f; unsigned u; } cv; cv.f = f;
    unsigned u = cv.u;
    u += 0x7FFFu + ((u >> 16) & 1u);          // round-to-nearest-even
    return (unsigned short)(u >> 16);
}
__device__ __forceinline__ float bf2f(unsigned short h) {
    union { unsigned u; float f; } v; v.u = (unsigned)h << 16; return v.f;
}
__device__ __forceinline__ __bf16 f2bf(float f) {
    unsigned short h = f2bf_u16(f);
    __bf16 b;
    __builtin_memcpy(&b, &h, sizeof(b));
    return b;
}
// pack two fp32 -> one u32 holding 2 bf16 (RNE)
__device__ __forceinline__ unsigned packbf2(float lo, float hi) {
    union { float f; unsigned u; } a, b;
    a.f = lo; b.f = hi;
    unsigned ua = a.u + (0x7FFFu + ((a.u >> 16) & 1u));
    unsigned ub = b.u + (0x7FFFu + ((b.u >> 16) & 1u));
    return (ua >> 16) | (ub & 0xFFFF0000u);
}
__device__ __forceinline__ v8f wmma_bf16(v16bf a, v16bf b, v8f c) {
    return __builtin_amdgcn_wmma_f32_16x16x32_bf16(
        false, a, false, b, (short)0, c, false, false);
}
// direct bf16 fragment load: elems i -> p[hf*8+i], elems 8+i -> p[16+hf*8+i]
__device__ __forceinline__ void loadfrag(FragAB& F, const unsigned short* p, int hf) {
    F.q[0] = *(const uint4*)(p + hf * 8);
    F.q[1] = *(const uint4*)(p + 16 + hf * 8);
}

// async 16-byte global -> LDS copy (falls back to a plain copy if unavailable)
__device__ __forceinline__ void async_cp16(const unsigned short* g, unsigned short* l) {
#if HAS_ASYNC
    __builtin_amdgcn_global_load_async_to_lds_b128((as1_v4i*)g, (as3_v4i*)l, 0, 0);
#else
    *(uint4*)l = *(const uint4*)g;
#endif
}
// wait for this wave's staged tile (TDM path or async path)
__device__ __forceinline__ void wait_stage() {
#if HAS_TDM
    __builtin_amdgcn_s_wait_tensorcnt(0);
#elif HAS_ASYNC
    __builtin_amdgcn_s_wait_asynccnt(0);
#endif
}

// ======================================================================
// fp32 -> bf16 bulk conversion (convert once, stream bf16 in all GEMMs)
// ======================================================================
__global__ __launch_bounds__(256)
void convert_f32_bf16(const float* __restrict__ src, unsigned short* __restrict__ dst, int n8)
{
    int i = blockIdx.x * 256 + threadIdx.x;
    if (i >= n8) return;
    const float4* s = (const float4*)src + (size_t)i * 2;
    float4 lo = s[0], hi = s[1];
    uint4 o;
    o.x = packbf2(lo.x, lo.y); o.y = packbf2(lo.z, lo.w);
    o.z = packbf2(hi.x, hi.y); o.w = packbf2(hi.z, hi.w);
    ((uint4*)dst)[i] = o;
}

// ======================================================================
// Pure-bf16 GEMM: C[M,N] = A[M,K] @ B[N,K]^T, 16x64/wave, K pipelined by 64
// ======================================================================
template <bool F32OUT>
__global__ __launch_bounds__(128)
void gemm_bf16(const unsigned short* __restrict__ A, const unsigned short* __restrict__ B,
               void* __restrict__ Cout, int M, int N, int K)
{
    const int lane = threadIdx.x & 31;
    const int wv   = threadIdx.x >> 5;
    const int l16  = lane & 15;
    const int hf   = lane >> 4;
    const int mTile = blockIdx.x;
    const int nBase = (blockIdx.y * 4 + wv) * 64;
    if (nBase >= N || mTile * 16 >= M) return;

    const int row = mTile * 16 + l16;
    const unsigned short* arow = A + (size_t)row * K;
    const unsigned short* br[4];
    #pragma unroll
    for (int t = 0; t < 4; ++t)
        br[t] = B + (size_t)(nBase + t * 16 + l16) * K;

    v8f zero = {};
    v8f acc[4];
    #pragma unroll
    for (int t = 0; t < 4; ++t) acc[t] = zero;

    FragAB a0, a1, b0[4], b1[4];
    loadfrag(a0, arow, hf);
    #pragma unroll
    for (int t = 0; t < 4; ++t) loadfrag(b0[t], br[t], hf);

    for (int k0 = 0; k0 < K; k0 += 64) {
        if (k0 + 128 < K) {                       // prefetch 2 stages ahead
            __builtin_prefetch(arow + k0 + 128, 0, 3);
            __builtin_prefetch(br[0] + k0 + 128, 0, 3);
            __builtin_prefetch(br[2] + k0 + 128, 0, 3);
        }
        loadfrag(a1, arow + k0 + 32, hf);
        #pragma unroll
        for (int t = 0; t < 4; ++t) loadfrag(b1[t], br[t] + k0 + 32, hf);
        #pragma unroll
        for (int t = 0; t < 4; ++t) acc[t] = wmma_bf16(a0.v, b0[t].v, acc[t]);

        if (k0 + 64 < K) {
            loadfrag(a0, arow + k0 + 64, hf);
            #pragma unroll
            for (int t = 0; t < 4; ++t) loadfrag(b0[t], br[t] + k0 + 64, hf);
        }
        #pragma unroll
        for (int t = 0; t < 4; ++t) acc[t] = wmma_bf16(a1.v, b1[t].v, acc[t]);
    }

    #pragma unroll
    for (int t = 0; t < 4; ++t) {
        FragC r; r.v = acc[t];
        const int col = nBase + t * 16 + l16;
        #pragma unroll
        for (int i = 0; i < 8; ++i) {
            const size_t idx = (size_t)(mTile * 16 + i + hf * 8) * N + col;
            if (F32OUT) ((float*)Cout)[idx] = r.f[i];
            else        ((unsigned short*)Cout)[idx] = f2bf_u16(r.f[i]);
        }
    }
}

// ======================================================================
// fused RoPE (bf16 in / bf16 out) + (b,s,h,d) -> (b,h,s,d) relayout
// ======================================================================
__global__ __launch_bounds__(128)
void rope_convert(const unsigned short* __restrict__ src, unsigned short* __restrict__ dst,
                  int nheads)
{
    const int bh = blockIdx.x;
    const int s  = blockIdx.y;
    const int b  = bh / nheads, h = bh % nheads;
    const int i  = threadIdx.x;                 // 0..127 rotation pairs

    const unsigned short* p = src + ((size_t)(b * kS + s) * nheads + h) * kHD;
    float x1 = bf2f(p[i]), x2 = bf2f(p[i + kHD / 2]);
    float inv = __expf(-(2.0f * (float)i / (float)kHD) * kLnBase);
    float ang = (float)s * inv;
    float sn, cs; __sincosf(ang, &sn, &cs);
    float o1 = x1 * cs - x2 * sn;
    float o2 = x2 * cs + x1 * sn;
    unsigned short* q = dst + ((size_t)(b * nheads + h) * kS + s) * kHD;
    q[i]           = f2bf_u16(o1);
    q[i + kHD / 2] = f2bf_u16(o2);
}

// ======================================================================
// V bf16 (b,s,kv,d) -> bf16 transposed (b,kv,d,s)
// ======================================================================
__global__ __launch_bounds__(256)
void v_transpose(const unsigned short* __restrict__ src, unsigned short* __restrict__ dst)
{
    const int bh = blockIdx.x;                  // b*NKV + h
    const int s  = blockIdx.y;
    const int d  = threadIdx.x;                 // 0..255
    const int b  = bh / kNKV, h = bh % kNKV;
    dst[((size_t)bh * kHD + d) * kS + s] =
        src[((size_t)(b * kS + s) * kNKV + h) * kHD + d];
}

// ======================================================================
// Flash attention: 2 waves / block, 32 queries / block.
// K tiles (32 keys x 256d bf16 = 16KB contiguous) double-buffered in LDS,
// staged by the Tensor Data Mover (one TENSOR_LOAD_TO_LDS per tile, wave 0
// only, TENSORcnt), falling back to GLOBAL_LOAD_ASYNC_TO_LDS_B128.
// ======================================================================
__global__ __launch_bounds__(64)
void attn_kernel(const unsigned short* __restrict__ Q,
                 const unsigned short* __restrict__ Kt,
                 const unsigned short* __restrict__ Vt,
                 unsigned short* __restrict__ ctx)
{
    __shared__ __align__(16) unsigned short qtile[2][16 * kHD];  // 16 KB
    __shared__ __align__(16) unsigned short ktile[2][32 * kHD];  // 32 KB (double buffer)
    __shared__ __align__(16) float ptile[2][16 * 32];            // 4 KB

    const int bh  = blockIdx.x;                 // b*NH + h
    const int b   = bh / kNH, h = bh % kNH;
    const int kvh = h / kGR;
    const int tid  = threadIdx.x;
    const int lane = tid & 31;
    const int wv   = tid >> 5;
    const int l16  = lane & 15;
    const int hf   = lane >> 4;
    const int q0b  = blockIdx.y * 32;           // block's first query
    const int q0   = q0b + wv * 16;             // wave's first query

    const unsigned short* kbaseptr = Kt + (size_t)(b * kNKV + kvh) * kS * kHD;
    const unsigned short* vbaseptr = Vt + (size_t)(b * kNKV + kvh) * kHD * kS;

    // stage both Q tiles (32 rows x 256 bf16, contiguous) cooperatively
    {
        const uint4* qs = (const uint4*)(Q + ((size_t)bh * kS + q0b) * kHD);
        uint4* qd = (uint4*)&qtile[0][0];
        for (int t = tid; t < 32 * kHD / 8; t += 64) qd[t] = qs[t];
    }

    v8f zero = {};
    v8f oacc[16];
    #pragma unroll
    for (int c = 0; c < 16; ++c) oacc[c] = zero;
    float rmax[8], rsum[8];
    #pragma unroll
    for (int r = 0; r < 8; ++r) { rmax[r] = -3.0e38f; rsum[r] = 0.0f; }

    const int nkt_blk = (q0b + 31) / 32 + 1;    // tiles the block must stage
    const int nkt_w   = (q0 + 15) / 32 + 1;     // tiles this wave computes

    // stage one 16KB K tile into LDS buffer bufi
    auto stage_k = [&](int kt, int bufi) {
        const unsigned short* ksrc = kbaseptr + (size_t)(kt * 32) * kHD;
        unsigned short* kdst = &ktile[bufi][0];
#if HAS_TDM
        if (wv == 0) {
            // D# group 0: count=1 | lds_addr | global_addr[56:0] | type=2
            unsigned ldsa = (unsigned)(unsigned long long)(as3_u16*)kdst;
            unsigned long long ga = (unsigned long long)ksrc;
            tdm_u32x4 g0;
            g0[0] = 1u;
            g0[1] = ldsa;
            g0[2] = (unsigned)ga;
            g0[3] = (unsigned)((ga >> 32) & 0x1FFFFFFull) | (2u << 30);
            // D# group 1: 1-row tile of 2048 x 8B elements, contiguous
            const unsigned elems = (32u * kHD * 2u) / 8u;   // 2048
            tdm_i32x8 g1 = {0, 0, 0, 0, 0, 0, 0, 0};
            g1[0] = (int)(3u << 16);                        // data_size = 8 bytes
            g1[1] = (int)((elems & 0xFFFFu) << 16);         // tensor_dim0[15:0]
            g1[2] = (int)(((elems >> 16) & 0xFFFFu) | (1u << 16)); // td0 hi | tensor_dim1=1
            g1[3] = (int)(elems << 16);                     // tile_dim0 = 2048
            g1[4] = 1;                                      // tile_dim1 = 1
            g1[5] = (int)elems;                             // tensor_dim0_stride
            tdm_i32x4 gz4 = {0, 0, 0, 0};
            tdm_i32x8 gz8 = {0, 0, 0, 0, 0, 0, 0, 0};
            __builtin_amdgcn_tensor_load_to_lds(g0, g1, gz4, gz4, gz8, 0);
        }
#else
        for (int t = tid; t < 1024; t += 64)
            async_cp16(ksrc + t * 8, kdst + t * 8);
#endif
    };

    stage_k(0, 0);
    __syncthreads();                            // covers qtile staging too

    for (int kt = 0; kt < nkt_blk; ++kt) {
        const int buf = kt & 1;
        wait_stage();
        __syncthreads();                        // K tile kt visible to both waves
        if (kt + 1 < nkt_blk) stage_k(kt + 1, buf ^ 1);

        if (kt < nkt_w) {
            const int kbase = kt * 32;
            const unsigned short* qt = &qtile[wv][0];
            const unsigned short* kbuf = &ktile[buf][0];
            float pv[2][8];

            // ---- scores: two 16-key halves, HD swept in 32-chunks ----
            #pragma unroll
            for (int hh = 0; hh < 2; ++hh) {
                const unsigned short* krow = kbuf + (size_t)(hh * 16 + l16) * kHD;
                v8f sacc = zero;
                #pragma unroll
                for (int d0 = 0; d0 < kHD; d0 += 32) {
                    FragAB A, Bf;
                    A.q[0]  = *(const uint4*)(qt + l16 * kHD + d0 + hf * 8);
                    A.q[1]  = *(const uint4*)(qt + l16 * kHD + d0 + 16 + hf * 8);
                    Bf.q[0] = *(const uint4*)(krow + d0 + hf * 8);
                    Bf.q[1] = *(const uint4*)(krow + d0 + 16 + hf * 8);
                    sacc = wmma_bf16(A.v, Bf.v, sacc);
                }
                FragC sr; sr.v = sacc;
                #pragma unroll
                for (int r = 0; r < 8; ++r) {
                    const int qrow = q0 + r + hf * 8;
                    const int kcol = kbase + hh * 16 + l16;
                    pv[hh][r] = sr.f[r] * kScale + (kcol > qrow ? -1.0e9f : 0.0f);
                }
            }

            // ---- online softmax (row stats via 16-lane xor reductions) ----
            float alphav[8];
            #pragma unroll
            for (int r = 0; r < 8; ++r) {
                float m = fmaxf(pv[0][r], pv[1][r]);
                #pragma unroll
                for (int off = 1; off < 16; off <<= 1)
                    m = fmaxf(m, __shfl_xor(m, off, 32));
                const float nmax = fmaxf(rmax[r], m);
                const float e0 = __expf(pv[0][r] - nmax);
                const float e1 = __expf(pv[1][r] - nmax);
                pv[0][r] = e0; pv[1][r] = e1;
                float s = e0 + e1;
                #pragma unroll
                for (int off = 1; off < 16; off <<= 1)
                    s += __shfl_xor(s, off, 32);
                const float alpha = __expf(rmax[r] - nmax);
                rsum[r] = rsum[r] * alpha + s;
                rmax[r] = nmax;
                alphav[r] = alpha;
            }
            #pragma unroll
            for (int c = 0; c < 16; ++c) {
                FragC oc; oc.v = oacc[c];
                #pragma unroll
                for (int r = 0; r < 8; ++r) oc.f[r] *= alphav[r];
                oacc[c] = oc.v;
            }

            // ---- P: C layout -> A layout through per-wave LDS tile ----
            float* pt = &ptile[wv][0];
            #pragma unroll
            for (int hh = 0; hh < 2; ++hh)
                #pragma unroll
                for (int r = 0; r < 8; ++r)
                    pt[(r + hf * 8) * 32 + hh * 16 + l16] = pv[hh][r];
            __threadfence_block();              // per-wave DS is in-order; order the compiler
            FragAB Pa;
            #pragma unroll
            for (int i = 0; i < 8; ++i) {
                Pa.b[i]     = f2bf(pt[l16 * 32 + hf * 8 + i]);
                Pa.b[8 + i] = f2bf(pt[l16 * 32 + 16 + hf * 8 + i]);
            }

            // ---- O += P @ V (V^T rows are contiguous 16B global loads) ----
            #pragma unroll
            for (int c = 0; c < 16; ++c) {
                const unsigned short* vrow = vbaseptr + (size_t)(c * 16 + l16) * kS + kbase;
                FragAB Bf;
                Bf.q[0] = *(const uint4*)(vrow + hf * 8);
                Bf.q[1] = *(const uint4*)(vrow + 16 + hf * 8);
                oacc[c] = wmma_bf16(Pa.v, Bf.v, oacc[c]);
            }
        }
        __syncthreads();                        // buf reusable only after all waves done
    }

    // ---- epilogue: normalize, write ctx (b,s,H) bf16 ----
    #pragma unroll
    for (int c = 0; c < 16; ++c) {
        FragC oc; oc.v = oacc[c];
        #pragma unroll
        for (int i = 0; i < 8; ++i) {
            const int srow = q0 + i + hf * 8;
            const int d    = c * 16 + l16;
            ctx[(size_t)(b * kS + srow) * kH + h * kHD + d] =
                f2bf_u16(oc.f[i] / rsum[i]);
        }
    }
}

// ======================================================================
// host launcher
// ======================================================================
extern "C" void kernel_launch(void* const* d_in, const int* in_sizes, int n_in,
                              void* d_out, int out_size, void* d_ws, size_t ws_size,
                              hipStream_t stream)
{
    (void)in_sizes; (void)n_in; (void)out_size; (void)ws_size;
    const float* hidden = (const float*)d_in[0];
    // d_in[1] = attention_mask: causal, applied analytically (never read)
    const float* Wq = (const float*)d_in[2];
    const float* Wk = (const float*)d_in[3];
    const float* Wv = (const float*)d_in[4];
    const float* Wo = (const float*)d_in[5];
    float* out = (float*)d_out;

    char* ws = (char*)d_ws;
    size_t off = 0;
    auto carve = [&](size_t bytes) {
        char* p = ws + off;
        off += (bytes + 255) & ~(size_t)255;
        return p;
    };
    const int kNKVH = kNKV * kHD;                                 // 768
    unsigned short* xbf  = (unsigned short*)carve((size_t)kM * kH * 2);
    unsigned short* wqb  = (unsigned short*)carve((size_t)kH * kH * 2);
    unsigned short* wkb  = (unsigned short*)carve((size_t)kNKVH * kH * 2);
    unsigned short* wvb  = (unsigned short*)carve((size_t)kNKVH * kH * 2);
    unsigned short* wob  = (unsigned short*)carve((size_t)kH * kH * 2);
    unsigned short* qraw = (unsigned short*)carve((size_t)kM * kH * 2);    // reused as ctx
    unsigned short* kraw = (unsigned short*)carve((size_t)kM * kNKVH * 2);
    unsigned short* vraw = (unsigned short*)carve((size_t)kM * kNKVH * 2);
    unsigned short* qbf  = (unsigned short*)carve((size_t)kM * kH * 2);
    unsigned short* kbf  = (unsigned short*)carve((size_t)kM * kNKVH * 2);
    unsigned short* vtb  = (unsigned short*)carve((size_t)kM * kNKVH * 2);
    unsigned short* ctx  = qraw;                                  // alias: dead after RoPE

    // 0) convert activations + weights to bf16 once (reused 15-256x by GEMMs)
    auto conv = [&](const float* s, unsigned short* d, size_t n) {
        int n8 = (int)(n / 8);
        convert_f32_bf16<<<dim3((n8 + 255) / 256), 256, 0, stream>>>(s, d, n8);
    };
    conv(hidden, xbf, (size_t)kM * kH);
    conv(Wq, wqb, (size_t)kH * kH);
    conv(Wk, wkb, (size_t)kNKVH * kH);
    conv(Wv, wvb, (size_t)kNKVH * kH);
    conv(Wo, wob, (size_t)kH * kH);

    // 1) QKV projections: pure-bf16 WMMA GEMMs, bf16 epilogue
    gemm_bf16<false><<<dim3(kM / 16, kH / 256), 128, 0, stream>>>(xbf, wqb, qraw, kM, kH, kH);
    gemm_bf16<false><<<dim3(kM / 16, kNKVH / 256), 128, 0, stream>>>(xbf, wkb, kraw, kM, kNKVH, kH);
    gemm_bf16<false><<<dim3(kM / 16, kNKVH / 256), 128, 0, stream>>>(xbf, wvb, vraw, kM, kNKVH, kH);

    // 2) RoPE + relayout to attention-friendly bf16 layouts
    rope_convert<<<dim3(kB * kNH, kS), 128, 0, stream>>>(qraw, qbf, kNH);
    rope_convert<<<dim3(kB * kNKV, kS), 128, 0, stream>>>(kraw, kbf, kNKV);
    v_transpose<<<dim3(kB * kNKV, kS), 256, 0, stream>>>(vraw, vtb);

    // 3) flash attention (GQA, causal), TDM K staging -> ctx bf16
    attn_kernel<<<dim3(kB * kNH, kS / 32), 64, 0, stream>>>(qbf, kbf, vtb, ctx);

    // 4) output projection -> fp32
    gemm_bf16<true><<<dim3(kM / 16, kH / 256), 128, 0, stream>>>(ctx, wob, out, kM, kH, kH);
}
